// ST_BLOCK_3_40888088658282
// MI455X (gfx1250) — compile-verified, matrix operand order checked by reading, not run on hardware
//
#include <hip/hip_runtime.h>
#include <math.h>
#include <stdint.h>

#define BZ   16
#define CIN  3
#define FDIM 64
#define NN   307
#define NP   320
#define TT   24
#define BT   (BZ*TT)      // 384
#define ATTLD 328         // padded att row stride (halfs)
#define HLD   328         // padded h row stride in LDS (halfs)

// dynamic LDS partition (bytes)
#define SM_HBUF   0                               // [2][64][HLD] f16 = 83968
#define SM_ATTB   (SM_HBUF + 2*64*HLD*2)          // [2][16][ATTLD] f16 = 20992
#define SM_EBUF   (SM_ATTB + 2*16*ATTLD*2)        // [16][ATTLD] f32 = 20992
#define SM_XSH    (SM_EBUF + 16*ATTLD*4)          // [2][16][64] f32 = 8192
#define SM_SD     (SM_XSH + 2*16*FDIM*4)          // [2][320] f32 = 2560
#define SM_TOTAL  (SM_SD + 2*NP*4)                // 136704

typedef _Float16 v8h_t  __attribute__((ext_vector_type(8)));
typedef _Float16 v16h_t __attribute__((ext_vector_type(16)));
typedef float    v8f_t  __attribute__((ext_vector_type(8)));

__device__ __forceinline__ v16h_t mk16(v8h_t lo, v8h_t hi) {
    union { v16h_t v; v8h_t p[2]; } u;
    u.p[0] = lo; u.p[1] = hi;
    return u.v;
}
// Fragment loader: 8 contiguous halfs at +koff, 8 more at +koff+16.
// CDNA5 16-bit A (16x32)/B (32x16) lane layout: lanes 0-15 take K {0..7,16..23},
// lanes 16-31 take K {8..15,24..31} (koff = 0 or 8).
__device__ __forceinline__ v16h_t ldfrag(const _Float16* base, int koff) {
    v8h_t lo = *(const v8h_t*)(base + koff);
    v8h_t hi = *(const v8h_t*)(base + koff + 16);
    return mk16(lo, hi);
}

__device__ __forceinline__ float lrelu(float x, float s) { return x > 0.f ? x : s * x; }

// ---------------------------------------------------------------------------
// K0: temporal conv (1,3) pad (0,1) + bias + lrelu -> xt f16 [bt][320][64]
// ---------------------------------------------------------------------------
__global__ __launch_bounds__(256) void k_prep(const float* __restrict__ x,
                                              const float* __restrict__ tw,
                                              const float* __restrict__ tb,
                                              _Float16* __restrict__ xt) {
    int idx = blockIdx.x * 256 + threadIdx.x;
    if (idx >= BT * NP) return;
    int n = idx % NP, bt = idx / NP;
    int b = bt / TT, t = bt % TT;
    _Float16* out = xt + (size_t)idx * FDIM;
    if (n >= NN) { for (int o = 0; o < FDIM; o++) out[o] = (_Float16)0.f; return; }
    float xin[CIN][3];
    for (int ci = 0; ci < CIN; ci++)
        for (int k = 0; k < 3; k++) {
            int ts = t + k - 1;
            xin[ci][k] = (ts >= 0 && ts < TT)
                ? x[(((size_t)b * CIN + ci) * NN + n) * TT + ts] : 0.f;
        }
    for (int o = 0; o < FDIM; o++) {
        float acc = tb[o];
        for (int ci = 0; ci < CIN; ci++)
            for (int k = 0; k < 3; k++)
                acc += xin[ci][k] * tw[(o * CIN + ci) * 3 + k];
        out[o] = (_Float16)lrelu(acc, 0.01f);
    }
}

// ---------------------------------------------------------------------------
// K0b: W1/W2 (f32 [c][f]) -> f16 transposed [which][f][c]
// ---------------------------------------------------------------------------
__global__ __launch_bounds__(256) void k_wt(const float* __restrict__ W1,
                                            const float* __restrict__ W2,
                                            _Float16* __restrict__ wt) {
    int idx = blockIdx.x * 256 + threadIdx.x;
    if (idx >= 2 * FDIM * FDIM) return;
    int which = idx / (FDIM * FDIM), r = idx % (FDIM * FDIM);
    int f = r / FDIM, c = r % FDIM;
    const float* W = which ? W2 : W1;
    wt[idx] = (_Float16)W[c * FDIM + f];
}

// ---------------------------------------------------------------------------
// K1: h = xt @ W via WMMA. 8 waves = {gat 0/1} x {ftile 0..3}.
// Output f16 F-major: hg[gat][bt][f][320] (each (gat,bt) tile is one
// contiguous 40960B region -> async-stageable as a linear burst in k_gat)
// ---------------------------------------------------------------------------
__global__ __launch_bounds__(256) void k_gemm1(const _Float16* __restrict__ xt,
                                               const _Float16* __restrict__ wt,
                                               _Float16* __restrict__ hg) {
    int mtile = blockIdx.x, bt = blockIdx.y;
    int lane = threadIdx.x & 31, wv = threadIdx.x >> 5;
    int gat = wv >> 2, ftile = wv & 3;
    int row  = mtile * 16 + (lane & 15);
    int koff = (lane < 16) ? 0 : 8;
    const _Float16* abase = xt + ((size_t)bt * NP + row) * FDIM;
    int f = ftile * 16 + (lane & 15);
    const _Float16* bbase = wt + (size_t)gat * FDIM * FDIM + (size_t)f * FDIM;
    v8f_t acc = {0.f, 0.f, 0.f, 0.f, 0.f, 0.f, 0.f, 0.f};
    for (int ks = 0; ks < 2; ks++) {   // K = 64 = 2 x 32
        v16h_t A = ldfrag(abase + ks * 32, koff);
        v16h_t B = ldfrag(bbase + ks * 32, koff);
        acc = __builtin_amdgcn_wmma_f32_16x16x32_f16(false, A, false, B,
                                                     (short)0, acc, false, false);
    }
    _Float16* hb = hg + (size_t)gat * BT * FDIM * NP
                      + ((size_t)bt * FDIM + f) * NP + mtile * 16;
    int mb = (lane < 16) ? 0 : 8;
    for (int r = 0; r < 8; r++) hb[mb + r] = (_Float16)acc[r];
}

// ---------------------------------------------------------------------------
// K1b: attention projections s = h . a  (4 dots in one pass over h1,h2)
// sarr layout: [4][bt][320] = {s1_src, s1_dst, s2_src, s2_dst}
// ---------------------------------------------------------------------------
__global__ __launch_bounds__(256) void k_sdots(const _Float16* __restrict__ hg,
                                               const float* __restrict__ a1s,
                                               const float* __restrict__ a1d,
                                               const float* __restrict__ a2s,
                                               const float* __restrict__ a2d,
                                               float* __restrict__ sarr) {
    int idx = blockIdx.x * 256 + threadIdx.x;
    if (idx >= BT * NP) return;
    int bt = idx / NP, n = idx % NP;
    const _Float16* h1 = hg + (size_t)bt * FDIM * NP + n;
    const _Float16* h2 = h1 + (size_t)BT * FDIM * NP;
    float s0 = 0, s1 = 0, s2 = 0, s3 = 0;
    for (int f = 0; f < FDIM; f++) {
        float v1 = (float)h1[(size_t)f * NP];
        float v2 = (float)h2[(size_t)f * NP];
        s0 += v1 * a1s[f]; s1 += v1 * a1d[f];
        s2 += v2 * a2s[f]; s3 += v2 * a2d[f];
    }
    size_t stride = (size_t)BT * NP;
    sarr[idx] = s0; sarr[stride + idx] = s1;
    sarr[2 * stride + idx] = s2; sarr[3 * stride + idx] = s3;
}

// ---------------------------------------------------------------------------
// K2: fused GAT. Per block: (bt, 16-row tile).
//   - issue async DMA of both 64x320 h tiles into LDS at kernel entry
//   - masked leaky-softmax per row (overlapped with the async DMA)
//   - s_wait_asynccnt 0 + barrier, then att@h via WMMA from LDS
//   - combine sigmoid(gate)*lrelu(filt)
// ---------------------------------------------------------------------------
__global__ __launch_bounds__(256) void k_gat(const _Float16* __restrict__ hg,
                                             const float* __restrict__ sarr,
                                             const float* __restrict__ supports,
                                             float* __restrict__ Sout,
                                             float* __restrict__ x1out) {
    extern __shared__ __align__(16) char smem[];
    _Float16* hbuf = (_Float16*)(smem + SM_HBUF);   // [2][64][HLD]
    _Float16* attb = (_Float16*)(smem + SM_ATTB);   // [2][16][ATTLD]
    float*    ebuf = (float*)   (smem + SM_EBUF);   // [16][ATTLD]
    float*    xsh  = (float*)   (smem + SM_XSH);    // [2][16][64]
    float*    sd   = (float*)   (smem + SM_SD);     // [2][320]

    int mtile = blockIdx.x, bt = blockIdx.y;
    int b = bt / TT, t = bt % TT;
    int tid = threadIdx.x, lane = tid & 31, wv = tid >> 5;
    size_t sstride = (size_t)BT * NP;

    // --- async-stage both h tiles (2 x 64 rows x 640B) into LDS, overlapped
    // with the softmax stage below. 5120 b128 chunks = 20 per thread.
    {
        uint64_t g0 = (uint64_t)(uintptr_t)(hg + (size_t)bt * FDIM * NP);
        uint64_t g1 = (uint64_t)(uintptr_t)(hg + (size_t)BT * FDIM * NP
                                               + (size_t)bt * FDIM * NP);
        uint32_t lb = (uint32_t)(uintptr_t)hbuf;   // low 32 bits = LDS offset
        for (int i = tid; i < 5120; i += 256) {
            int gat = i / 2560;
            int r   = i % 2560;              // 64 rows * 40 chunks
            int row = r / 40, c = r % 40;
            uint32_t ld = lb + (uint32_t)(gat * (64 * HLD * 2)
                                          + row * (HLD * 2) + c * 16);
            uint64_t ga = (gat ? g1 : g0) + (uint64_t)(row * (NP * 2) + c * 16);
            asm volatile("global_load_async_to_lds_b128 %0, %1, off"
                         :: "v"(ld), "v"(ga) : "memory");
        }
    }

    // stage A: stage dst projections into LDS
    for (int i = tid; i < 2 * NP; i += 256) {
        int g = i / NP, j = i % NP;
        sd[g * NP + j] = sarr[(size_t)(g ? 3 : 1) * sstride + (size_t)bt * NP + j];
    }
    __syncthreads();

    // stage B: masked leaky-softmax, one wave per row (2 rows/wave, 2 gats)
    for (int gat = 0; gat < 2; gat++) {
        const float* ssrc = sarr + (size_t)(gat ? 2 : 0) * sstride + (size_t)bt * NP;
        for (int rr = 0; rr < 2; rr++) {
            int i  = wv * 2 + rr;
            int ng = mtile * 16 + i;
            float* er = ebuf + i * ATTLD;
            _Float16* ar = attb + (size_t)gat * 16 * ATTLD + i * ATTLD;
            if (ng < NN) {
                float si = ssrc[ng];
                const float* adj = supports + (size_t)ng * NN;
                float mx = -3.0e38f;
                for (int j = lane; j < NP; j += 32) {
                    float e = -3.0e38f;
                    if (j < NN) {
                        float v = si + sd[gat * NP + j];
                        v = lrelu(v, 0.3f);
                        e = (adj[j] > 0.f) ? v : -9.0e15f;
                    }
                    er[j] = e;
                    mx = fmaxf(mx, e);
                }
                for (int o = 16; o; o >>= 1) mx = fmaxf(mx, __shfl_xor(mx, o, 32));
                float sum = 0.f;
                for (int j = lane; j < NP; j += 32) {
                    float ex = (j < NN) ? __expf(er[j] - mx) : 0.f;
                    er[j] = ex; sum += ex;
                }
                for (int o = 16; o; o >>= 1) sum += __shfl_xor(sum, o, 32);
                float inv = 1.f / sum;
                for (int j = lane; j < NP; j += 32) {
                    float a = er[j] * inv;
                    ar[j] = (_Float16)a;
                    if (gat == 0 && j < NN)
                        Sout[((size_t)bt * NN + ng) * NN + j] = a;   // S_coef
                }
            } else {
                for (int j = lane; j < NP; j += 32) ar[j] = (_Float16)0.f;
            }
        }
    }

    // wait for the async DMA of h tiles, then make them block-visible
    asm volatile("s_wait_asynccnt 0x0" ::: "memory");
    __syncthreads();

    // stage C: (16x320 att) @ (320x16 h) per wave via 10 chained WMMAs (LDS)
    {
        int gat = wv >> 2, ftile = wv & 3;
        int koff = (lane < 16) ? 0 : 8;
        const _Float16* arow = attb + (size_t)gat * 16 * ATTLD + (lane & 15) * ATTLD;
        int f = ftile * 16 + (lane & 15);
        const _Float16* bbase = hbuf + (size_t)gat * 64 * HLD + (size_t)f * HLD;
        v8f_t acc = {0.f, 0.f, 0.f, 0.f, 0.f, 0.f, 0.f, 0.f};
        for (int ks = 0; ks < 10; ks++) {  // K = 320
            v16h_t A = ldfrag(arow + ks * 32, koff);
            v16h_t B = ldfrag(bbase + ks * 32, koff);
            acc = __builtin_amdgcn_wmma_f32_16x16x32_f16(false, A, false, B,
                                                         (short)0, acc, false, false);
        }
        int mb = (lane < 16) ? 0 : 8;
        for (int r = 0; r < 8; r++) xsh[gat * 16 * FDIM + (mb + r) * FDIM + f] = acc[r];
    }
    __syncthreads();

    // stage D: gating, write x1 [b][f][n][t]
    for (int i = tid; i < 16 * FDIM; i += 256) {
        int m = i / FDIM, f = i % FDIM;
        int ng = mtile * 16 + m;
        if (ng < NN) {
            float fl = xsh[i], gt = xsh[16 * FDIM + i];
            float sg = 1.f / (1.f + __expf(-gt));
            x1out[(((size_t)b * FDIM + f) * NN + ng) * TT + t] = sg * lrelu(fl, 0.01f);
        }
    }
}

// ---------------------------------------------------------------------------
// TATT small kernels
// ---------------------------------------------------------------------------
__global__ __launch_bounds__(256) void k_f1(const float* __restrict__ x1,
                                            const float* __restrict__ t1w,
                                            float* __restrict__ f1) {
    int idx = blockIdx.x * 256 + threadIdx.x;
    if (idx >= BT * NN) return;
    int bt = idx / NN, n = idx % NN;
    int b = bt / TT, t = bt % TT;
    float acc = 0.f;
    for (int c = 0; c < FDIM; c++)
        acc += x1[(((size_t)b * FDIM + c) * NN + n) * TT + t] * t1w[c];
    f1[idx] = acc;
}

__global__ __launch_bounds__(256) void k_f2(const float* __restrict__ x1,
                                            const float* __restrict__ t2w,
                                            float* __restrict__ f2) {
    int idx = blockIdx.x * 256 + threadIdx.x;
    if (idx >= BZ * FDIM * TT) return;
    int b = idx / (FDIM * TT), r = idx % (FDIM * TT);
    int c = r / TT, t = r % TT;
    float acc = 0.f;
    const float* xr = x1 + (((size_t)b * FDIM + c) * NN) * TT + t;
    for (int n = 0; n < NN; n++) acc += xr[(size_t)n * TT] * t2w[n];
    f2[idx] = acc;
}

__global__ __launch_bounds__(256) void k_g(const float* __restrict__ f1,
                                           const float* __restrict__ tw,
                                           float* __restrict__ g) {
    int idx = blockIdx.x * 256 + threadIdx.x;
    if (idx >= BT * FDIM) return;
    int bt = idx / FDIM, c = idx % FDIM;
    float acc = 0.f;
    const float* fr = f1 + (size_t)bt * NN;
    for (int n = 0; n < NN; n++) acc += fr[n] * tw[n * FDIM + c];
    g[idx] = acc;
}

__global__ __launch_bounds__(256) void k_tatt(const float* __restrict__ g,
                                              const float* __restrict__ f2,
                                              const float* __restrict__ tb,
                                              const float* __restrict__ tv,
                                              const float* __restrict__ bng,
                                              const float* __restrict__ bnb,
                                              float* __restrict__ coefs,
                                              float* __restrict__ Tout) {
    __shared__ float lb[TT * TT], l2[TT * TT];
    int b = blockIdx.x, tid = threadIdx.x;
    for (int i = tid; i < TT * TT; i += 256) {
        int t = i / TT, q = i % TT;
        float acc = tb[i];
        const float* gr  = g  + ((size_t)b * TT + t) * FDIM;
        const float* f2c = f2 + (size_t)b * FDIM * TT;
        for (int c = 0; c < FDIM; c++) acc += gr[c] * f2c[c * TT + q];
        lb[i] = 1.f / (1.f + __expf(-acc));
    }
    __syncthreads();
    const float inv = 0.9999950000374997f;  // 1/sqrt(1+1e-5)
    for (int i = tid; i < TT * TT; i += 256) {
        int p = i / TT, q = i % TT;
        float acc = 0.f;
        for (int t = 0; t < TT; t++) acc += tv[p * TT + t] * lb[t * TT + q];
        acc = acc * inv * bng[q] + bnb[q];
        if (q > p) acc += -1e13f;            // causal mask
        l2[i] = acc;
    }
    __syncthreads();
    if (tid < TT) {
        int p = tid;
        float mx = -3e38f;
        for (int q = 0; q < TT; q++) mx = fmaxf(mx, l2[p * TT + q]);
        float ex[TT], s = 0.f;
        for (int q = 0; q < TT; q++) { ex[q] = __expf(l2[p * TT + q] - mx); s += ex[q]; }
        float invs = 1.f / s;
        for (int q = 0; q < TT; q++) {
            float c = ex[q] * invs;
            coefs[((size_t)b * TT + p) * TT + q] = c;   // coefs[b][p][q]
            Tout [((size_t)b * TT + q) * TT + p] = c;   // T_coef = transpose
        }
    }
}

// ---------------------------------------------------------------------------
// K4: apply T_coef + residual (1x1 conv recomputed from x) + stats atomics
// ---------------------------------------------------------------------------
__global__ __launch_bounds__(256) void k_combine(const float* __restrict__ x1,
                                                 const float* __restrict__ coefs,
                                                 const float* __restrict__ x,
                                                 const float* __restrict__ c1w,
                                                 const float* __restrict__ c1b,
                                                 float* __restrict__ yout,
                                                 float* __restrict__ stats) {
    __shared__ float cf[TT * TT];
    __shared__ float rs[8][2];
    int b = blockIdx.y;
    for (int i = threadIdx.x; i < TT * TT; i += 256)
        cf[i] = coefs[(size_t)b * TT * TT + i];
    __syncthreads();
    int idx = blockIdx.x * 256 + threadIdx.x;
    float lsum = 0.f, lsq = 0.f;
    if (idx < FDIM * NN) {
        int c = idx / NN, n = idx % NN;
        const float* xr = x1 + (((size_t)b * FDIM + c) * NN + n) * TT;
        float xv[TT];
        for (int l = 0; l < TT; l++) xv[l] = xr[l];
        float w0 = c1w[c * CIN + 0], w1 = c1w[c * CIN + 1], w2 = c1w[c * CIN + 2];
        float bb = c1b[c];
        const float* x0 = x + (((size_t)b * CIN + 0) * NN + n) * TT;
        const float* xA = x + (((size_t)b * CIN + 1) * NN + n) * TT;
        const float* xB = x + (((size_t)b * CIN + 2) * NN + n) * TT;
        float* yr = yout + (((size_t)b * FDIM + c) * NN + n) * TT;
        for (int q = 0; q < TT; q++) {
            float acc = 0.f;
            for (int l = 0; l < TT; l++) acc += xv[l] * cf[q * TT + l];
            float y = lrelu(acc, 0.01f)
                    + bb + w0 * x0[q] + w1 * xA[q] + w2 * xB[q];
            yr[q] = y;
            lsum += y; lsq += y * y;
        }
    }
    for (int o = 16; o; o >>= 1) { lsum += __shfl_xor(lsum, o, 32); lsq += __shfl_xor(lsq, o, 32); }
    int lane = threadIdx.x & 31, wv = threadIdx.x >> 5;
    if (!lane) { rs[wv][0] = lsum; rs[wv][1] = lsq; }
    __syncthreads();
    if (threadIdx.x == 0) {
        float a = 0.f, s = 0.f;
        for (int w = 0; w < 8; w++) { a += rs[w][0]; s += rs[w][1]; }
        atomicAdd(&stats[b * 2], a);
        atomicAdd(&stats[b * 2 + 1], s);
    }
}

__global__ void k_stats(const float* __restrict__ stats, float* __restrict__ mr) {
    int b = threadIdx.x;
    if (b < BZ) {
        const float invn = 1.f / (float)(FDIM * NN * TT);
        float mu  = stats[b * 2] * invn;
        float var = stats[b * 2 + 1] * invn - mu * mu;
        mr[b * 2] = mu;
        mr[b * 2 + 1] = rsqrtf(var + 1e-5f);
    }
}

__global__ __launch_bounds__(256) void k_ln(float* __restrict__ out,
                                            const float* __restrict__ mr,
                                            const float* __restrict__ lnw,
                                            const float* __restrict__ lnb) {
    size_t idx = (size_t)blockIdx.x * 256 + threadIdx.x;
    const size_t per = (size_t)FDIM * NN * TT;
    if (idx >= (size_t)BZ * per) return;
    int b = (int)(idx / per);
    size_t r = idx % per;
    float y = out[idx];
    out[idx] = (y - mr[b * 2]) * mr[b * 2 + 1] * lnw[r] + lnb[r];
}

// ---------------------------------------------------------------------------
extern "C" void kernel_launch(void* const* d_in, const int* in_sizes, int n_in,
                              void* d_out, int out_size, void* d_ws, size_t ws_size,
                              hipStream_t stream) {
    const float* x        = (const float*)d_in[0];
    const float* supports = (const float*)d_in[1];
    const float* conv1_w  = (const float*)d_in[2];
    const float* conv1_b  = (const float*)d_in[3];
    const float* time_w   = (const float*)d_in[4];
    const float* time_b   = (const float*)d_in[5];
    const float* W1       = (const float*)d_in[6];
    const float* a1s      = (const float*)d_in[7];
    const float* a1d      = (const float*)d_in[8];
    const float* W2       = (const float*)d_in[9];
    const float* a2s      = (const float*)d_in[10];
    const float* a2d      = (const float*)d_in[11];
    const float* t1w      = (const float*)d_in[12];
    const float* t2w      = (const float*)d_in[13];
    const float* tw       = (const float*)d_in[14];
    const float* tb       = (const float*)d_in[15];
    const float* tv       = (const float*)d_in[16];
    const float* bn_g     = (const float*)d_in[17];
    const float* bn_b     = (const float*)d_in[18];
    const float* ln_w     = (const float*)d_in[19];
    const float* ln_b     = (const float*)d_in[20];

    float* out0 = (float*)d_out;                       // [16,64,307,24]
    float* Sout = out0 + (size_t)BZ * FDIM * NN * TT;  // [16,24,307,307]
    float* Tout = Sout + (size_t)BZ * TT * NN * NN;    // [16,24,24]

    char* ws = (char*)d_ws;
    size_t off = 0;
    auto carve = [&](size_t bytes) { size_t o = off; off += (bytes + 255) & ~(size_t)255; return o; };
    _Float16* xt   = (_Float16*)(ws + carve((size_t)BT * NP * FDIM * 2));
    _Float16* wt   = (_Float16*)(ws + carve((size_t)2 * FDIM * FDIM * 2));
    _Float16* hg   = (_Float16*)(ws + carve((size_t)2 * BT * FDIM * NP * 2));
    float*    sarr = (float*)   (ws + carve((size_t)4 * BT * NP * 4));
    float*    x1b  = (float*)   (ws + carve((size_t)BZ * FDIM * NN * TT * 4));
    float*    f1   = (float*)   (ws + carve((size_t)BT * NN * 4));
    float*    f2   = (float*)   (ws + carve((size_t)BZ * FDIM * TT * 4));
    float*    g    = (float*)   (ws + carve((size_t)BT * FDIM * 4));
    float*    coef = (float*)   (ws + carve((size_t)BZ * TT * TT * 4));
    float*    stats= (float*)   (ws + carve((size_t)BZ * 2 * 4));
    float*    mr   = (float*)   (ws + carve((size_t)BZ * 2 * 4));

    // allow >64KB dynamic LDS for the fused GAT kernel (WGP has 320KB)
    (void)hipFuncSetAttribute(reinterpret_cast<const void*>(k_gat),
                              hipFuncAttributeMaxDynamicSharedMemorySize,
                              SM_TOTAL);

    k_prep <<<(BT * NP + 255) / 256, 256, 0, stream>>>(x, time_w, time_b, xt);
    k_wt   <<<(2 * FDIM * FDIM + 255) / 256, 256, 0, stream>>>(W1, W2, wt);
    k_gemm1<<<dim3(20, BT), 256, 0, stream>>>(xt, wt, hg);
    k_sdots<<<(BT * NP + 255) / 256, 256, 0, stream>>>(hg, a1s, a1d, a2s, a2d, sarr);
    k_gat  <<<dim3(20, BT), 256, SM_TOTAL, stream>>>(hg, sarr, supports, Sout, x1b);
    k_f1   <<<(BT * NN + 255) / 256, 256, 0, stream>>>(x1b, t1w, f1);
    k_f2   <<<(BZ * FDIM * TT + 255) / 256, 256, 0, stream>>>(x1b, t2w, f2);
    k_g    <<<(BT * FDIM + 255) / 256, 256, 0, stream>>>(f1, tw, g);
    k_tatt <<<BZ, 256, 0, stream>>>(g, f2, tb, tv, bn_g, bn_b, coef, Tout);
    hipMemsetAsync(stats, 0, (size_t)BZ * 2 * 4, stream);
    k_combine<<<dim3((FDIM * NN + 255) / 256, BZ), 256, 0, stream>>>(
        x1b, coef, x, conv1_w, conv1_b, out0, stats);
    k_stats<<<1, 32, 0, stream>>>(stats, mr);
    k_ln   <<<((int)((size_t)BZ * FDIM * NN * TT + 255) / 256), 256, 0, stream>>>(
        out0, mr, ln_w, ln_b);
}
